// NodeModel_62766652064044
// MI455X (gfx1250) — matrix operand order repaired and checked by reading, not compiled, not run on hardware
//
#include <hip/hip_runtime.h>

typedef __attribute__((ext_vector_type(16))) __bf16 v16bf;
typedef __attribute__((ext_vector_type(8)))  float  v8f;

#define KDW   48     // 96 features packed 2 bf16 per dword
#define PITCH 50     // padded LDS row pitch in dwords (100 halves) -> conflict-free frag loads
#define WAVES 4
#define TILE_M 16
#define MAX_BLOCKS 2048

// LDS layout (dword offsets)
#define WT1_S 0                         // 96 * PITCH
#define WT2_S (WT1_S + 96 * PITCH)      // 64 * PITCH
#define ACT_S (WT2_S + 64 * PITCH)      // WAVES * 16 * PITCH
#define HID_S (ACT_S + WAVES * 16 * PITCH)
#define SMEM_DW (HID_S + WAVES * 16 * PITCH)   // 14400 dwords = 57.6 KB

// Round-half-up f32->bf16 pair pack: 2x v_add_nc_u32 + 1x v_perm_b32.
__device__ __forceinline__ unsigned int pack2_bf16(float a, float b) {
  union { float f; unsigned int u; } ua, ub;
  ua.f = a; ub.f = b;
  unsigned int ra = ua.u + 0x8000u;
  unsigned int rb = ub.u + 0x8000u;
  // result = { rb[31:16], ra[31:16] }  (a in low half)
  return __builtin_amdgcn_perm(rb, ra, 0x07060302);
}

// Single-instruction ReLU (fmaxf(x,0) canonicalizes into 2 v_max ops otherwise).
__device__ __forceinline__ float relu1(float x) {
  float r;
  asm("v_max_num_f32 %0, %1, 0" : "=v"(r) : "v"(x));
  return r;
}

__device__ __forceinline__ v8f wmma_bf16(v16bf a, v16bf b, v8f c) {
  return __builtin_amdgcn_wmma_f32_16x16x32_bf16(false, a, false, b, (short)0, c,
                                                 false, false);
}

// A fragment (16x32 bf16, MxK): lane&15 = row M; lanes 16-31 -> K base +8.
// VGPR v (0..3): K = h*8 + 2v ; VGPR v (4..7): K = 16 + h*8 + 2(v-4).
__device__ __forceinline__ v16bf load_a(const unsigned int* base, int lane, int ks) {
  union { unsigned int u[8]; v16bf v; } f;
  const unsigned int* p = base + (lane & 15) * PITCH + ks * 16 + ((lane >> 4) << 2);
#pragma unroll
  for (int v = 0; v < 4; ++v) { f.u[v] = p[v]; f.u[v + 4] = p[v + 8]; }
  return f.v;
}

// B fragment (32x16 bf16, KxN), weights stored transposed: row n, K contiguous.
// VGPR v (0..3): K = h*8 + 2v ; VGPR v (4..7): K = 16 + h*8 + 2(v-4).
__device__ __forceinline__ v16bf load_b(const unsigned int* base, int lane, int ks) {
  union { unsigned int u[8]; v16bf v; } f;
  const unsigned int* p = base + (lane & 15) * PITCH + ks * 16 + ((lane >> 4) << 2);
#pragma unroll
  for (int v = 0; v < 4; ++v) { f.u[v] = p[v]; f.u[v + 4] = p[v + 8]; }
  return f.v;
}

__global__ void zero_kernel(float* __restrict__ out, float* __restrict__ counts,
                            int n_out, int n_nodes) {
  int stride = gridDim.x * blockDim.x;
  for (int i = blockIdx.x * blockDim.x + threadIdx.x; i < n_out; i += stride)
    out[i] = 0.0f;
  for (int i = blockIdx.x * blockDim.x + threadIdx.x; i < n_nodes; i += stride)
    counts[i] = 0.0f;
}

// Transpose + bf16-pack W1 (96x96) and W2 (96x64) into K-pair dwords.
__global__ void prep_weights(const float* __restrict__ W1, const float* __restrict__ W2,
                             unsigned int* __restrict__ wt1g,
                             unsigned int* __restrict__ wt2g) {
  int idx = blockIdx.x * blockDim.x + threadIdx.x;
  if (idx < 96 * KDW) {
    int n = idx / KDW, d = idx % KDW;
    wt1g[idx] = pack2_bf16(W1[(2 * d) * 96 + n], W1[(2 * d + 1) * 96 + n]);
  }
  idx -= 96 * KDW;
  if (idx >= 0 && idx < 64 * KDW) {
    int n = idx / KDW, d = idx % KDW;
    wt2g[idx] = pack2_bf16(W2[(2 * d) * 64 + n], W2[(2 * d + 1) * 64 + n]);
  }
}

__global__ __launch_bounds__(WAVES * 32)
void edge_mlp_kernel(const float* __restrict__ x, const int* __restrict__ ei,
                     const float* __restrict__ ea, const float* __restrict__ b1,
                     const float* __restrict__ b2,
                     const unsigned int* __restrict__ wt1g,
                     const unsigned int* __restrict__ wt2g,
                     float* __restrict__ out, float* __restrict__ counts,
                     int n_edges) {
  __shared__ unsigned int smem[SMEM_DW];
  const int tid  = threadIdx.x;
  const int lane = tid & 31;
  const int wv   = tid >> 5;

  // Cooperative weight staging into LDS (bf16, transposed, padded rows). Once per block.
  for (int i = tid; i < 96 * KDW; i += WAVES * 32)
    smem[WT1_S + (i / KDW) * PITCH + (i % KDW)] = wt1g[i];
  for (int i = tid; i < 64 * KDW; i += WAVES * 32)
    smem[WT2_S + (i / KDW) * PITCH + (i % KDW)] = wt2g[i];
  __syncthreads();

  unsigned int*   actw  = smem + ACT_S + wv * 16 * PITCH;
  unsigned int*   hidw  = smem + HID_S + wv * 16 * PITCH;
  unsigned short* hid16 = (unsigned short*)hidw;
  const int ncol  = lane & 15;
  const int mbase = (lane >> 4) * 8;

  const int n_tiles    = (n_edges + TILE_M - 1) / TILE_M;
  const int wave_gid   = blockIdx.x * WAVES + wv;
  const int wave_count = gridDim.x * WAVES;

  // Persistent waves: each wave grid-strides over 16-edge tiles.
  // acts/hidden regions are wave-private -> no barrier inside the loop.
  for (int t = wave_gid; t < n_tiles; t += wave_count) {
    const int eb = t * TILE_M;

    // Stage concat(x[row[e]], edge_attr[e]) as bf16.
    // 2 lanes per edge row; float4 global loads + b64 LDS stores.
    {
      int m = lane >> 1, sub = lane & 1;
      int e = eb + m; if (e >= n_edges) e = n_edges - 1;
      int src = ei[e];
      const float4* xr = (const float4*)(x + (long long)src * 64);
      const float4* er = (const float4*)(ea + (long long)e * 32);
      unsigned int* arow = actw + m * PITCH;
#pragma unroll
      for (int jj = 0; jj < 12; ++jj) {
        int d0 = sub * 24 + jj * 2;                 // even dword index in 0..46
        const float4* p = (d0 < 32) ? (xr + (d0 >> 1)) : (er + ((d0 - 32) >> 1));
        float4 f = *p;
        uint2 pr;
        pr.x = pack2_bf16(f.x, f.y);
        pr.y = pack2_bf16(f.z, f.w);
        *(uint2*)(arow + d0) = pr;                  // 8B-aligned (d0 even, PITCH even)
      }
    }
    // Per-edge in-degree counts (scatter-mean denominator)
    if (lane < 16 && (eb + lane) < n_edges) {
      int c = ei[n_edges + eb + lane];
      __hip_atomic_fetch_add(counts + c, 1.0f, __ATOMIC_RELAXED,
                             __HIP_MEMORY_SCOPE_AGENT);
    }

    // ---- Layer 1: h1 = relu(A @ W1 + b1), A: 16x96, W1: 96x96 ----
    v16bf A0 = load_a(actw, lane, 0);
    v16bf A1 = load_a(actw, lane, 1);
    v16bf A2 = load_a(actw, lane, 2);
#pragma unroll
    for (int nt = 0; nt < 6; ++nt) {
      float bv = b1[nt * 16 + ncol];
      v8f acc = {bv, bv, bv, bv, bv, bv, bv, bv};
      const unsigned int* wb = smem + WT1_S + nt * 16 * PITCH;
      acc = wmma_bf16(A0, load_b(wb, lane, 0), acc);
      acc = wmma_bf16(A1, load_b(wb, lane, 1), acc);
      acc = wmma_bf16(A2, load_b(wb, lane, 2), acc);
#pragma unroll
      for (int r = 0; r < 8; ++r) {
        union { float f; unsigned int u; } c;
        c.f = relu1(acc[r]);
        unsigned int rb = c.u + 0x8000u;            // round-half-up to bf16 in [31:16]
        hid16[(mbase + r) * (2 * PITCH) + nt * 16 + ncol] =
            (unsigned short)(rb >> 16);             // folds into ds_store_b16_d16_hi
      }
    }

    // ---- Layer 2: h2 = h1 @ W2 + b2, then scatter-add ----
    v16bf H0 = load_a(hidw, lane, 0);
    v16bf H1 = load_a(hidw, lane, 1);
    v16bf H2 = load_a(hidw, lane, 2);
    int cdst[8];
#pragma unroll
    for (int r = 0; r < 8; ++r) {
      int e = eb + mbase + r;
      cdst[r] = (e < n_edges) ? ei[n_edges + e] : -1;
    }
#pragma unroll
    for (int nt = 0; nt < 4; ++nt) {
      float bv = b2[nt * 16 + ncol];
      v8f acc = {bv, bv, bv, bv, bv, bv, bv, bv};
      const unsigned int* wb = smem + WT2_S + nt * 16 * PITCH;
      acc = wmma_bf16(H0, load_b(wb, lane, 0), acc);
      acc = wmma_bf16(H1, load_b(wb, lane, 1), acc);
      acc = wmma_bf16(H2, load_b(wb, lane, 2), acc);
#pragma unroll
      for (int r = 0; r < 8; ++r) {
        if (cdst[r] >= 0) {
          __hip_atomic_fetch_add(out + (long long)cdst[r] * 64 + nt * 16 + ncol,
                                 acc[r], __ATOMIC_RELAXED, __HIP_MEMORY_SCOPE_AGENT);
        }
      }
    }
  }
}

__global__ void finalize_kernel(float* __restrict__ out,
                                const float* __restrict__ counts, int n_out) {
  int stride = gridDim.x * blockDim.x;
  for (int i = blockIdx.x * blockDim.x + threadIdx.x; i < n_out; i += stride) {
    float c = counts[i >> 6];
    out[i] = out[i] / fmaxf(c, 1.0f);
  }
}

extern "C" void kernel_launch(void* const* d_in, const int* in_sizes, int n_in,
                              void* d_out, int out_size, void* d_ws, size_t ws_size,
                              hipStream_t stream) {
  const float* x   = (const float*)d_in[0];
  const int*   ei  = (const int*)d_in[1];   // [2, E] int32 (jax default x64-disabled)
  const float* ea  = (const float*)d_in[2];
  const float* W1  = (const float*)d_in[3];
  const float* b1  = (const float*)d_in[4];
  const float* W2  = (const float*)d_in[5];
  const float* b2  = (const float*)d_in[6];
  float* out = (float*)d_out;

  int n_nodes = in_sizes[0] / 64;
  int n_edges = in_sizes[2] / 32;
  int n_out   = n_nodes * 64;

  // Workspace: counts (n_nodes f32, padded) | wt1g (96*48 dw) | wt2g (64*48 dw)
  float* counts = (float*)d_ws;
  size_t cnt_dw = (size_t)((n_nodes + 255) & ~255);
  unsigned int* wt1g = (unsigned int*)d_ws + cnt_dw;
  unsigned int* wt2g = wt1g + 96 * KDW;

  zero_kernel<<<512, 256, 0, stream>>>(out, counts, n_out, n_nodes);
  prep_weights<<<(96 * KDW + 64 * KDW + 255) / 256, 256, 0, stream>>>(W1, W2, wt1g, wt2g);

  int n_tiles = (n_edges + TILE_M - 1) / TILE_M;
  int nblocks = (n_tiles + WAVES - 1) / WAVES;
  if (nblocks > MAX_BLOCKS) nblocks = MAX_BLOCKS;
  edge_mlp_kernel<<<nblocks, WAVES * 32, 0, stream>>>(x, ei, ea, b1, b2, wt1g, wt2g,
                                                      out, counts, n_edges);
  finalize_kernel<<<512, 256, 0, stream>>>(out, counts, n_out);
}
